// GeeArrYou_16295105921441
// MI455X (gfx1250) — matrix operand order repaired and checked by reading, not compiled, not run on hardware
//
#include <hip/hip_runtime.h>

// ---------------------------------------------------------------------------
// Persistent-weight GRU for MI455X (gfx1250, wave32, WMMA).
//
//   gates = [h_t, x_t] @ [W_hh, W_ih]^T          (bf16 WMMA, f32 accum)
//   r = sig(.), z = sig(.), n = tanh(gi_n + r*gh_n), h' = (1-z)n + z h
//
// 32 persistent workgroups, each owns 16 hidden units (48 gate rows).
// Weights converted f32->bf16 once into LDS; h double-buffered in global
// (f32 for the z*h term + bf16 mirror for async staging); one atomic grid
// barrier per timestep.  Staging uses global_load_async_to_lds_b128.
// Kernel is templated on KTOT so all WMMA loops fully unroll.
// ---------------------------------------------------------------------------

#define HID   512
#define BATCH 16
#define SEQ   2048
#define NWG   32          // hidden slice per WG = 512/32 = 16
#define TPB   256         // 8 waves; waves 0..5 run WMMA

typedef __attribute__((ext_vector_type(8)))  float  v8f;
typedef __attribute__((ext_vector_type(8)))  __bf16 v8bf;
typedef __attribute__((ext_vector_type(16))) __bf16 v16bf;

__device__ __forceinline__ unsigned f2bf(float f) {      // RNE f32 -> bf16 bits
  unsigned u = __float_as_uint(f);
  return (u + 0x7FFFu + ((u >> 16) & 1u)) >> 16;
}

__device__ __forceinline__ v8f wmma_bf16(v16bf a, v16bf b, v8f c) {
  return __builtin_amdgcn_wmma_f32_16x16x32_bf16(false, a, false, b, (short)0, c,
                                                 false, false);
}

// CDNA5 async copy: 16B global -> LDS, tracked by ASYNCcnt (no VGPR data path)
__device__ __forceinline__ void async_g2l_b128(unsigned lds_off,
                                               unsigned long long gaddr) {
  asm volatile("global_load_async_to_lds_b128 %0, %1, off"
               :: "v"(lds_off), "v"(gaddr) : "memory");
}
__device__ __forceinline__ void wait_async0() {
  asm volatile("s_wait_asynccnt 0" ::: "memory");
}

__device__ __forceinline__ float fast_sigmoid(float x) {
  return __builtin_amdgcn_rcpf(1.0f + __expf(-x));       // v_exp + v_rcp
}
__device__ __forceinline__ float fast_tanh(float x) {
  // tanh(x) = 1 - 2/(e^{2x}+1); saturates correctly as exp -> 0 / inf
  return 1.0f - 2.0f * __builtin_amdgcn_rcpf(__expf(2.0f * x) + 1.0f);
}

__device__ __forceinline__ void grid_barrier(unsigned* cnt, unsigned* gen) {
  __syncthreads();
  if (threadIdx.x == 0) {
    unsigned g = __hip_atomic_load(gen, __ATOMIC_RELAXED, __HIP_MEMORY_SCOPE_AGENT);
    unsigned a = __hip_atomic_fetch_add(cnt, 1u, __ATOMIC_ACQ_REL, __HIP_MEMORY_SCOPE_AGENT);
    if (a == NWG - 1u) {
      __hip_atomic_store(cnt, 0u, __ATOMIC_RELAXED, __HIP_MEMORY_SCOPE_AGENT);
      __hip_atomic_store(gen, g + 1u, __ATOMIC_RELEASE, __HIP_MEMORY_SCOPE_AGENT);
    } else {
      while (__hip_atomic_load(gen, __ATOMIC_ACQUIRE, __HIP_MEMORY_SCOPE_AGENT) == g)
        __builtin_amdgcn_s_sleep(2);
    }
  }
  __syncthreads();
}

template <int KTOT>
__global__ __launch_bounds__(TPB, 1)
void gru_layer(const float* __restrict__ Whh, const float* __restrict__ Wih,
               const float* __restrict__ bih, const float* __restrict__ bhh,
               const float* __restrict__ xf32,          // layer0 input (f32) or null
               const unsigned short* __restrict__ xbf,  // layer1 input (bf16 hseq) or null
               float* __restrict__ hbuf0, float* __restrict__ hbuf1,
               unsigned short* __restrict__ hb0,        // bf16 mirrors of h state
               unsigned short* __restrict__ hb1,
               unsigned* __restrict__ bar_cnt, unsigned* __restrict__ bar_gen,
               unsigned short* __restrict__ hseq_out,   // layer0 output (bf16) or null
               float* __restrict__ out_f32,             // layer1 output (f32) or null
               int in_dim)
{
  constexpr int STRIDE = KTOT + 8;      // halves; +16B row pad to spread LDS banks
  constexpr int TT     = KTOT >> 5;     // 32-wide k-tiles (24 or 32)

  const int wg  = blockIdx.x;
  const int j0  = wg * 16;              // hidden slice owned by this WG
  const int tid = threadIdx.x;

  extern __shared__ __align__(16) char smem[];
  unsigned short* Wl = (unsigned short*)smem;        // [48][STRIDE] bf16 weights
  unsigned short* Al = Wl + 48 * STRIDE;             // [16][STRIDE] bf16 [h_t, x_t]
  float* sH    = (float*)(Al + 16 * STRIDE);         // [2 khalf][3 gate][16][16]
  float* sX    = sH + 2 * 3 * 256;                   // same shape (x-part sums)
  float* biasC = sX + 2 * 3 * 256;                   // [32]  combined r,z biases
  float* bihn  = biasC + 32;                         // [16]
  float* bhhn  = bihn + 16;                          // [16]
  const unsigned AlBase = (unsigned)(uintptr_t)Al;   // LDS byte offset of A tile

  // ---- one-time: convert this WG's 48 gate rows of [W_hh | W_ih] to bf16 LDS
  for (int lr = 0; lr < 48; ++lr) {
    int gate = lr >> 4, jj = lr & 15;
    int g = gate * HID + j0 + jj;
    for (int k = tid; k < KTOT; k += TPB) {
      float v = (k < HID) ? Whh[g * HID + k] : Wih[g * in_dim + (k - HID)];
      Wl[lr * STRIDE + k] = (unsigned short)f2bf(v);
    }
  }
  if (tid < 16) {
    int jj = tid;
    biasC[jj]      = bih[j0 + jj]       + bhh[j0 + jj];            // r
    biasC[16 + jj] = bih[HID + j0 + jj] + bhh[HID + j0 + jj];      // z
    bihn[jj]       = bih[2 * HID + j0 + jj];
    bhhn[jj]       = bhh[2 * HID + j0 + jj];
  }
  { // zero h0 slice (f32 + bf16 mirror) in the t=0 read buffers
    int b = tid >> 4, jj = tid & 15;
    hbuf0[b * HID + j0 + jj] = 0.0f;
    hb0[b * HID + j0 + jj]   = 0;
  }
  grid_barrier(bar_cnt, bar_gen);

  // wave-uniform role: force into SGPRs so WMMA loops get scalar control flow
  const int wv   = __builtin_amdgcn_readfirstlane(tid >> 5);
  const int lane = tid & 31;
  const int m = lane & 15, hs = lane >> 4;

  for (int t = 0; t < SEQ; ++t) {
    const float* hR          = (t & 1) ? hbuf1 : hbuf0;
    float*       hW          = (t & 1) ? hbuf0 : hbuf1;
    const unsigned short* bR = (t & 1) ? hb1 : hb0;
    unsigned short*       bW = (t & 1) ? hb0 : hb1;

    // ---- phase 1: stage A = [h_t, x_t] (bf16) into LDS
    #pragma unroll
    for (int i = tid; i < 1024; i += TPB) {            // 16*512 halves / 8
      int b = i >> 6, c8 = (i & 63) << 3;              // 8-half chunks
      async_g2l_b128(AlBase + (unsigned)(b * STRIDE + c8) * 2u,
                     (unsigned long long)(uintptr_t)(bR + b * HID + c8));
    }
    if (xf32) {                                        // layer0: x f32 -> bf16
      const float4* x4 = (const float4*)xf32;
      #pragma unroll
      for (int i = tid; i < 16 * 64; i += TPB) {       // 1024 float4
        int b = i >> 6, kq = (i & 63) << 2;
        float4 v = x4[(b * SEQ + t) * 64 + (i & 63)];
        uint2 p;
        p.x = f2bf(v.x) | (f2bf(v.y) << 16);
        p.y = f2bf(v.z) | (f2bf(v.w) << 16);
        *(uint2*)(Al + b * STRIDE + HID + kq) = p;
      }
    } else {                                           // layer1: async copy hseq
      #pragma unroll
      for (int i = tid; i < 1024; i += TPB) {
        int b = i >> 6, c8 = (i & 63) << 3;
        async_g2l_b128(AlBase + (unsigned)(b * STRIDE + HID + c8) * 2u,
                       (unsigned long long)(uintptr_t)(xbf + (t * 16 + b) * HID + c8));
      }
    }
    wait_async0();
    __syncthreads();

    // ---- phase 2: WMMA GEMM.  6 waves = 3 gates x 2 K-interleave lanes.
    // Wave (gate, kh) handles k-tiles kt = kh, kh+2, ... -> constant trip
    // counts so both loops fully unroll with immediate ds_load offsets.
    if (wv < 6) {
      int gate = wv >> 1, kh = wv & 1;
      v8f ch = {};
      v8f cx = {};
      const unsigned short* arow = Al + m * STRIDE;
      const unsigned short* brow = Wl + (gate * 16 + m) * STRIDE;
      #pragma unroll
      for (int u = 0; u < 8; ++u) {                    // h-part: 16 tiles / 2
        int kt = kh + 2 * u;
        int ka = (kt << 5) + (hs << 3);
        v8bf a0 = *(const v8bf*)(arow + ka);
        v8bf a1 = *(const v8bf*)(arow + ka + 16);
        v16bf a = __builtin_shufflevector(a0, a1, 0,1,2,3,4,5,6,7,8,9,10,11,12,13,14,15);
        int kb = (kt << 5) + (hs << 3);
        v8bf b0 = *(const v8bf*)(brow + kb);
        v8bf b1 = *(const v8bf*)(brow + kb + 16);
        v16bf b = __builtin_shufflevector(b0, b1, 0,1,2,3,4,5,6,7,8,9,10,11,12,13,14,15);
        ch = wmma_bf16(a, b, ch);
      }
      #pragma unroll
      for (int u = 0; u < (TT - 16) / 2; ++u) {        // x-part tiles
        int kt = 16 + kh + 2 * u;
        int ka = (kt << 5) + (hs << 3);
        v8bf a0 = *(const v8bf*)(arow + ka);
        v8bf a1 = *(const v8bf*)(arow + ka + 16);
        v16bf a = __builtin_shufflevector(a0, a1, 0,1,2,3,4,5,6,7,8,9,10,11,12,13,14,15);
        int kb = (kt << 5) + (hs << 3);
        v8bf b0 = *(const v8bf*)(brow + kb);
        v8bf b1 = *(const v8bf*)(brow + kb + 16);
        v16bf b = __builtin_shufflevector(b0, b1, 0,1,2,3,4,5,6,7,8,9,10,11,12,13,14,15);
        cx = wmma_bf16(a, b, cx);
      }
      // D layout: element r <-> (M = r + (lane>=16 ? 8:0), N = lane&15)
      float* dh = sH + (kh * 3 + gate) * 256;
      float* dx = sX + (kh * 3 + gate) * 256;
      int mm = (lane < 16) ? 0 : 8;
      #pragma unroll
      for (int r = 0; r < 8; ++r) {
        dh[(r + mm) * 16 + m] = ch[r];
        dx[(r + mm) * 16 + m] = cx[r];
      }
    }
    __syncthreads();

    // ---- phase 3: gate nonlinearities + state update (1 elem / thread)
    {
      int b = tid >> 4, jj = tid & 15;
      int e = b * 16 + jj;
      float ghr = sH[0 * 256 + e] + sH[3 * 256 + e];
      float ghz = sH[1 * 256 + e] + sH[4 * 256 + e];
      float ghn = sH[2 * 256 + e] + sH[5 * 256 + e] + bhhn[jj];
      float gxr = sX[0 * 256 + e] + sX[3 * 256 + e];
      float gxz = sX[1 * 256 + e] + sX[4 * 256 + e];
      float gxn = sX[2 * 256 + e] + sX[5 * 256 + e] + bihn[jj];
      float r = fast_sigmoid(ghr + gxr + biasC[jj]);
      float z = fast_sigmoid(ghz + gxz + biasC[16 + jj]);
      float n = fast_tanh(gxn + r * ghn);
      float ho = hR[b * HID + j0 + jj];
      float hn = (1.0f - z) * n + z * ho;
      unsigned short hnb = (unsigned short)f2bf(hn);
      hW[b * HID + j0 + jj] = hn;
      bW[b * HID + j0 + jj] = hnb;
      if (hseq_out) hseq_out[(t * 16 + b) * HID + j0 + jj] = hnb;
      if (out_f32)  out_f32[((size_t)b * SEQ + t) * HID + j0 + jj] = hn;
    }

    grid_barrier(bar_cnt, bar_gen);   // publish h_{t+1} to all 32 WGPs
  }
}

extern "C" void kernel_launch(void* const* d_in, const int* in_sizes, int n_in,
                              void* d_out, int out_size, void* d_ws, size_t ws_size,
                              hipStream_t stream) {
  const float* x    = (const float*)d_in[0];
  const float* Wih0 = (const float*)d_in[1];
  const float* Whh0 = (const float*)d_in[2];
  const float* bih0 = (const float*)d_in[3];
  const float* bhh0 = (const float*)d_in[4];
  const float* Wih1 = (const float*)d_in[5];
  const float* Whh1 = (const float*)d_in[6];
  const float* bih1 = (const float*)d_in[7];
  const float* bhh1 = (const float*)d_in[8];
  float* out = (float*)d_out;

  char* ws = (char*)d_ws;
  unsigned* bar = (unsigned*)ws;                       // [0]=cnt, [1]=gen
  float* hbuf0 = (float*)(ws + 256);                   // 2 x 16*512 f32
  float* hbuf1 = hbuf0 + BATCH * HID;
  unsigned short* hb0 = (unsigned short*)(ws + 256 + 2 * BATCH * HID * sizeof(float));
  unsigned short* hb1 = hb0 + BATCH * HID;             // 2 x 16*512 bf16
  unsigned short* hseq = hb1 + BATCH * HID;            // 16*2048*512 bf16 (~32MB)

  hipMemsetAsync(ws, 0, 256, stream);                  // reset barrier state

  auto shbytes = [](int Ktot) -> size_t {
    size_t stride = (size_t)Ktot + 8;
    return 128 * stride /* (48+16) rows * stride halves * 2B */
         + (2 * 3 * 256 * 2 + 64) * sizeof(float);
  };

  constexpr int K0 = HID + 256, K1 = HID + HID;
  gru_layer<K0><<<dim3(NWG), dim3(TPB), shbytes(K0), stream>>>(
      Whh0, Wih0, bih0, bhh0, x, nullptr, hbuf0, hbuf1, hb0, hb1,
      bar, bar + 1, hseq, nullptr, 256);
  gru_layer<K1><<<dim3(NWG), dim3(TPB), shbytes(K1), stream>>>(
      Whh1, Wih1, bih1, bhh1, nullptr, hseq, hbuf0, hbuf1, hb0, hb1,
      bar, bar + 1, nullptr, out, HID);
}
